// ViTQuadraticAttention_62440234549860
// MI455X (gfx1250) — compile-verified, matrix operand order checked by reading, not run on hardware
//
#include <hip/hip_runtime.h>
#include <hip/hip_bf16.h>
#include <math.h>

// ---------------------------------------------------------------------------
// ViT quadratic attention, fused, bf16-WMMA (gfx1250 / CDNA5, wave32)
// ---------------------------------------------------------------------------

typedef __attribute__((ext_vector_type(16))) __bf16 bf16x16;
typedef __attribute__((ext_vector_type(8)))  __bf16 bf16x8;
typedef __attribute__((ext_vector_type(8)))  float  f32x8;

#define BATCH 8
#define NSEQ  1024
#define HS    768
#define NH    12
#define HD    64

union FragAB { bf16x16 v; bf16x8 h[2]; };

// Load a 16x32 bf16 A-fragment (or B-fragment, i.e. 16 contiguous-K "rows")
// per the CDNA5 16-bit matrix layout:
//   lanes 0-15 : M = lane,     K = [0..7] and [16..23]
//   lanes 16-31: M = lane-16,  K = [8..15] and [24..31]
__device__ __forceinline__ FragAB load_frag(const __bf16* __restrict__ base,
                                            int ldk, int row_base, int k0, int lane) {
    int half = lane >> 4;
    int r    = lane & 15;
    const __bf16* p = base + (size_t)(row_base + r) * (size_t)ldk + (size_t)(k0 + half * 8);
    FragAB f;
    f.h[0] = *(const bf16x8*)(p);
    f.h[1] = *(const bf16x8*)(p + 16);
    return f;
}

__device__ __forceinline__ f32x8 wmma_bf16(const FragAB& a, const FragAB& b, f32x8 c) {
    return __builtin_amdgcn_wmma_f32_16x16x32_bf16(
        /*neg_a=*/false, a.v, /*neg_b=*/false, b.v,
        /*c_mod=*/(short)0, c, /*reuse_a=*/false, /*reuse_b=*/false);
}

// ---------------------------------------------------------------------------
// Kernel 1: fp32 -> bf16 conversion (4-wide)
// ---------------------------------------------------------------------------
__global__ void cvt_f32_bf16(const float* __restrict__ src, __bf16* __restrict__ dst, int n4) {
    int i = blockIdx.x * blockDim.x + threadIdx.x;
    if (i >= n4) return;
    float4 f = ((const float4*)src)[i];
    __bf16* d = dst + (size_t)i * 4;
    d[0] = (__bf16)f.x; d[1] = (__bf16)f.y; d[2] = (__bf16)f.z; d[3] = (__bf16)f.w;
}

// ---------------------------------------------------------------------------
// Kernel 2: projection GEMM  Y[m,o] = sum_k X[m,k]*W[o,k] (+ bias[o])
// One wave per 16x64 output tile (one head-slice wide): the A fragment is
// reused across 4 B fragments / 4 accumulators -> 10 16B loads per 4 WMMAs.
// Output written straight into head-split layout:
//   transpose_out==0 -> [B, H, N, 64] (qd, kd, q, k)
//   transpose_out==1 -> [B, H, 64, N] (v, so P@V B-fragments are contiguous)
// ---------------------------------------------------------------------------
__global__ void __launch_bounds__(256) proj_gemm(
    const __bf16* __restrict__ X,   // [8192, 768]
    const __bf16* __restrict__ W,   // [768, 768]
    const float*  __restrict__ bias,// [768] or nullptr
    __bf16* __restrict__ out,
    int transpose_out)
{
    int lane = threadIdx.x & 31;
    int wave = threadIdx.x >> 5;
    int wid  = blockIdx.x * (blockDim.x >> 5) + wave;
    const int TILES_NQ = HS / 64;             // 12 column-quads (== heads)
    const int TILES_M  = (BATCH * NSEQ) / 16; // 512
    int tm = wid / TILES_NQ;
    int tq = wid % TILES_NQ;                  // head index
    if (tm >= TILES_M) return;
    int row_base = tm * 16;
    int col_base = tq * 64;

    f32x8 acc[4] = {};
    #pragma unroll 2
    for (int k0 = 0; k0 < HS; k0 += 32) {
        FragAB a = load_frag(X, HS, row_base, k0, lane);
        #pragma unroll
        for (int c = 0; c < 4; ++c) {
            FragAB b = load_frag(W, HS, col_base + c * 16, k0, lane);
            acc[c] = wmma_bf16(a, b, acc[c]);
        }
    }

    int half = lane >> 4;
    int r    = lane & 15;
    int h    = tq;                     // 64-wide quad == one head

    #pragma unroll
    for (int c = 0; c < 4; ++c) {
        int d0   = c * 16 + r;         // within-head feature
        float bv = bias ? bias[col_base + c * 16 + r] : 0.0f;
        #pragma unroll
        for (int e = 0; e < 8; ++e) {
            int m  = row_base + e + half * 8;   // C-layout rows e / e+8
            int b_ = m / NSEQ, n = m % NSEQ;
            float val = acc[c][e] + bv;
            size_t idx;
            if (transpose_out) idx = (((size_t)(b_ * NH + h)) * HD + d0) * NSEQ + n;
            else               idx = (((size_t)(b_ * NH + h)) * NSEQ + n) * HD + d0;
            out[idx] = (__bf16)val;
        }
    }
}

// ---------------------------------------------------------------------------
// Kernel 3: per-row sum of squares over head dim (for the -0.5*|.|^2 terms)
// ---------------------------------------------------------------------------
__global__ void rowsq(const __bf16* __restrict__ x, float* __restrict__ out, int rows) {
    int i = blockIdx.x * blockDim.x + threadIdx.x;
    if (i >= rows) return;
    const __bf16* p = x + (size_t)i * HD;
    float s = 0.f;
    #pragma unroll
    for (int v8 = 0; v8 < HD / 8; ++v8) {
        bf16x8 q = *(const bf16x8*)(p + v8 * 8);
        #pragma unroll
        for (int j = 0; j < 8; ++j) { float v = (float)q[j]; s += v * v; }
    }
    out[i] = s;
}

// ---------------------------------------------------------------------------
// Kernel 4: fused flash-style attention.
// One wave = one 16-row query tile of one (b,h). Loop over 32-key tiles:
//   8 WMMAs: S = qd*kd^T + q*k^T  (two 16-col halves, K=64 in 2 steps)
//   online softmax (shfl_xor row reductions within 16-lane halves)
//   P staged via LDS (C-layout -> A-layout), 4 WMMAs: O += P @ V
// Next key tile is prefetched (global_prefetch_b8) behind the current tile's
// matrix work.
// ---------------------------------------------------------------------------
__global__ void __launch_bounds__(256) attn_fused(
    const __bf16* __restrict__ qd, const __bf16* __restrict__ kd,
    const __bf16* __restrict__ qp, const __bf16* __restrict__ kp,
    const __bf16* __restrict__ vt,
    const float* __restrict__ qdsq, const float* __restrict__ kdsq,
    float* __restrict__ out)
{
    __shared__ __bf16 pstage[8][16 * 32];   // per-wave P tile, 1KB each

    int lane = threadIdx.x & 31;
    int wave = threadIdx.x >> 5;
    int wid  = blockIdx.x * 8 + wave;
    const int QT = NSEQ / 16;               // 64 query tiles per (b,h)
    int bh = wid / QT;
    int qt = wid % QT;
    if (bh >= BATCH * NH) return;
    int b = bh / NH, h = bh % NH;

    size_t base_nd = (size_t)bh * NSEQ * HD;
    size_t base_dn = (size_t)bh * HD * NSEQ;
    const __bf16* qd_b = qd + base_nd;
    const __bf16* kd_b = kd + base_nd;
    const __bf16* q_b  = qp + base_nd;
    const __bf16* k_b  = kp + base_nd;
    const __bf16* vt_b = vt + base_dn;      // [64, NSEQ]
    const float*  qsq_b = qdsq + (size_t)bh * NSEQ;
    const float*  ksq_b = kdsq + (size_t)bh * NSEQ;

    int half = lane >> 4;
    int r    = lane & 15;
    int qrow_base = qt * 16;

    // Hoisted A-fragments (query side), invariant across key tiles.
    FragAB aqd0 = load_frag(qd_b, HD, qrow_base, 0,  lane);
    FragAB aqd1 = load_frag(qd_b, HD, qrow_base, 32, lane);
    FragAB aq0  = load_frag(q_b,  HD, qrow_base, 0,  lane);
    FragAB aq1  = load_frag(q_b,  HD, qrow_base, 32, lane);

    float qsq[8];
    #pragma unroll
    for (int e = 0; e < 8; ++e) qsq[e] = qsq_b[qrow_base + e + half * 8];

    float mrow[8], lrow[8];
    f32x8 o0 = {}, o1 = {}, o2 = {}, o3 = {};
    #pragma unroll
    for (int e = 0; e < 8; ++e) { mrow[e] = -INFINITY; lrow[e] = 0.f; }

    __bf16* pw = pstage[wave];
    const float scale_s = 0.125f;   // 1/sqrt(64)

    for (int kt = 0; kt < NSEQ; kt += 32) {
        // ---- prefetch next key tile behind this tile's matrix work
        if (kt + 32 < NSEQ) {
            int nkt = kt + 32;
            // kd/k: 32 rows x 64 bf16 = 4KB contiguous; one lane per row
            __builtin_prefetch(kd_b + (size_t)(nkt + lane) * HD, 0, 3);
            __builtin_prefetch(k_b  + (size_t)(nkt + lane) * HD, 0, 3);
            // v^T: 64 strided rows, 64B segment each; two rows per lane
            __builtin_prefetch(vt_b + (size_t)lane        * NSEQ + nkt, 0, 3);
            __builtin_prefetch(vt_b + (size_t)(lane + 32) * NSEQ + nkt, 0, 3);
        }

        // ---- scores: quad + lin in one accumulator per 16-col half
        f32x8 s0 = {}, s1 = {};
        {
            FragAB bf;
            bf = load_frag(kd_b, HD, kt,      0,  lane); s0 = wmma_bf16(aqd0, bf, s0);
            bf = load_frag(kd_b, HD, kt,      32, lane); s0 = wmma_bf16(aqd1, bf, s0);
            bf = load_frag(k_b,  HD, kt,      0,  lane); s0 = wmma_bf16(aq0,  bf, s0);
            bf = load_frag(k_b,  HD, kt,      32, lane); s0 = wmma_bf16(aq1,  bf, s0);
            bf = load_frag(kd_b, HD, kt + 16, 0,  lane); s1 = wmma_bf16(aqd0, bf, s1);
            bf = load_frag(kd_b, HD, kt + 16, 32, lane); s1 = wmma_bf16(aqd1, bf, s1);
            bf = load_frag(k_b,  HD, kt + 16, 0,  lane); s1 = wmma_bf16(aq0,  bf, s1);
            bf = load_frag(k_b,  HD, kt + 16, 32, lane); s1 = wmma_bf16(aq1,  bf, s1);
        }

        float ksq0 = ksq_b[kt + r];
        float ksq1 = ksq_b[kt + 16 + r];

        float p0[8], p1[8], mt[8];
        #pragma unroll
        for (int e = 0; e < 8; ++e) {
            float v0 = (s0[e] - 0.5f * (qsq[e] + ksq0)) * scale_s;
            float v1 = (s1[e] - 0.5f * (qsq[e] + ksq1)) * scale_s;
            p0[e] = v0; p1[e] = v1;
            mt[e] = fmaxf(v0, v1);
        }
        // row max across the 16-lane half (C-layout row lives in 16 lanes)
        #pragma unroll
        for (int off = 1; off < 16; off <<= 1) {
            #pragma unroll
            for (int e = 0; e < 8; ++e)
                mt[e] = fmaxf(mt[e], __shfl_xor(mt[e], off, 32));
        }

        float scl[8];
        #pragma unroll
        for (int e = 0; e < 8; ++e) {
            float mnew = fmaxf(mrow[e], mt[e]);
            scl[e]  = __expf(mrow[e] - mnew);
            mrow[e] = mnew;
            p0[e] = __expf(p0[e] - mnew);
            p1[e] = __expf(p1[e] - mnew);
        }
        float rs[8];
        #pragma unroll
        for (int e = 0; e < 8; ++e) rs[e] = p0[e] + p1[e];
        #pragma unroll
        for (int off = 1; off < 16; off <<= 1) {
            #pragma unroll
            for (int e = 0; e < 8; ++e)
                rs[e] += __shfl_xor(rs[e], off, 32);
        }
        #pragma unroll
        for (int e = 0; e < 8; ++e) {
            lrow[e] = lrow[e] * scl[e] + rs[e];
            o0[e] *= scl[e]; o1[e] *= scl[e]; o2[e] *= scl[e]; o3[e] *= scl[e];
        }

        // ---- stage P (C-layout) to LDS, reload in A-layout
        #pragma unroll
        for (int e = 0; e < 8; ++e) {
            int row = e + half * 8;
            pw[row * 32 + r]      = (__bf16)p0[e];
            pw[row * 32 + 16 + r] = (__bf16)p1[e];
        }
        FragAB pa;
        {
            const __bf16* pp = pw + r * 32 + half * 8;
            pa.h[0] = *(const bf16x8*)(pp);
            pa.h[1] = *(const bf16x8*)(pp + 16);
        }

        // ---- O += P @ V  (V^T layout -> contiguous-key B-fragments)
        FragAB bv;
        bv = load_frag(vt_b, NSEQ, 0,  kt, lane); o0 = wmma_bf16(pa, bv, o0);
        bv = load_frag(vt_b, NSEQ, 16, kt, lane); o1 = wmma_bf16(pa, bv, o1);
        bv = load_frag(vt_b, NSEQ, 32, kt, lane); o2 = wmma_bf16(pa, bv, o2);
        bv = load_frag(vt_b, NSEQ, 48, kt, lane); o3 = wmma_bf16(pa, bv, o3);
    }

    // ---- normalize and store: out[b, n, h*64 + d], fp32
    float linv[8];
    #pragma unroll
    for (int e = 0; e < 8; ++e) linv[e] = 1.0f / lrow[e];
    #pragma unroll
    for (int e = 0; e < 8; ++e) {
        int n = qrow_base + e + half * 8;
        size_t ob = ((size_t)b * NSEQ + n) * HS + (size_t)h * HD;
        out[ob +  0 + r] = o0[e] * linv[e];
        out[ob + 16 + r] = o1[e] * linv[e];
        out[ob + 32 + r] = o2[e] * linv[e];
        out[ob + 48 + r] = o3[e] * linv[e];
    }
}

// ---------------------------------------------------------------------------
// Host-side launcher
// ---------------------------------------------------------------------------
extern "C" void kernel_launch(void* const* d_in, const int* in_sizes, int n_in,
                              void* d_out, int out_size, void* d_ws, size_t ws_size,
                              hipStream_t stream) {
    (void)in_sizes; (void)n_in; (void)out_size; (void)ws_size;

    const float* qs = (const float*)d_in[0];
    const float* ks = (const float*)d_in[1];
    const float* Wd = (const float*)d_in[2];
    const float* Wq = (const float*)d_in[3];
    const float* bq = (const float*)d_in[4];
    const float* Wk = (const float*)d_in[5];
    const float* bk = (const float*)d_in[6];
    const float* Wv = (const float*)d_in[7];
    const float* bv = (const float*)d_in[8];
    float* out = (float*)d_out;

    const size_t N_ACT = (size_t)BATCH * NSEQ * HS;   // 6291456
    const size_t N_W   = (size_t)HS * HS;             // 589824
    const size_t N_ROW = (size_t)BATCH * NH * NSEQ;   // 98304

    // workspace carve-up (bf16 buffers, 256B aligned)
    char* ws = (char*)d_ws;
    size_t cur = 0;
    auto carve = [&](size_t bytes) -> char* {
        char* p = ws + cur;
        cur += (bytes + 255) & ~(size_t)255;
        return p;
    };
    __bf16* qs_bf = (__bf16*)carve(N_ACT * 2);
    __bf16* ks_bf = (__bf16*)carve(N_ACT * 2);
    __bf16* wd_bf = (__bf16*)carve(N_W * 2);
    __bf16* wq_bf = (__bf16*)carve(N_W * 2);
    __bf16* wk_bf = (__bf16*)carve(N_W * 2);
    __bf16* wv_bf = (__bf16*)carve(N_W * 2);
    __bf16* qd_b  = (__bf16*)carve(N_ACT * 2);   // [B,H,N,64]
    __bf16* kd_b  = (__bf16*)carve(N_ACT * 2);
    __bf16* q_b   = (__bf16*)carve(N_ACT * 2);
    __bf16* k_b   = (__bf16*)carve(N_ACT * 2);
    __bf16* vt_b  = (__bf16*)carve(N_ACT * 2);   // [B,H,64,N]
    float*  qdsq  = (float*)carve(N_ROW * 4);
    float*  kdsq  = (float*)carve(N_ROW * 4);

    // 1) conversions
    {
        int n4a = (int)(N_ACT / 4), n4w = (int)(N_W / 4);
        cvt_f32_bf16<<<(n4a + 255) / 256, 256, 0, stream>>>(qs, qs_bf, n4a);
        cvt_f32_bf16<<<(n4a + 255) / 256, 256, 0, stream>>>(ks, ks_bf, n4a);
        cvt_f32_bf16<<<(n4w + 255) / 256, 256, 0, stream>>>(Wd, wd_bf, n4w);
        cvt_f32_bf16<<<(n4w + 255) / 256, 256, 0, stream>>>(Wq, wq_bf, n4w);
        cvt_f32_bf16<<<(n4w + 255) / 256, 256, 0, stream>>>(Wk, wk_bf, n4w);
        cvt_f32_bf16<<<(n4w + 255) / 256, 256, 0, stream>>>(Wv, wv_bf, n4w);
    }

    // 2) projections: 512 row-tiles * 12 head-quads = 6144 waves -> 768 blocks
    {
        dim3 grid(768), block(256);
        proj_gemm<<<grid, block, 0, stream>>>(qs_bf, wd_bf, nullptr, qd_b, 0);
        proj_gemm<<<grid, block, 0, stream>>>(ks_bf, wd_bf, nullptr, kd_b, 0);
        proj_gemm<<<grid, block, 0, stream>>>(qs_bf, wq_bf, bq,      q_b,  0);
        proj_gemm<<<grid, block, 0, stream>>>(ks_bf, wk_bf, bk,      k_b,  0);
        proj_gemm<<<grid, block, 0, stream>>>(ks_bf, wv_bf, bv,      vt_b, 1);
    }

    // 3) row |.|^2 for qd, kd
    rowsq<<<(int)((N_ROW + 255) / 256), 256, 0, stream>>>(qd_b, qdsq, (int)N_ROW);
    rowsq<<<(int)((N_ROW + 255) / 256), 256, 0, stream>>>(kd_b, kdsq, (int)N_ROW);

    // 4) fused attention: 96 (b,h) * 64 q-tiles = 6144 waves -> 768 blocks
    attn_fused<<<768, 256, 0, stream>>>(qd_b, kd_b, q_b, k_b, vt_b, qdsq, kdsq, out);
}
